// LogNCDE_28415503630798
// MI455X (gfx1250) — matrix operand order repaired and checked by reading, not compiled, not run on hardware
//
#include <hip/hip_runtime.h>
#include <hip/hip_bf16.h>
#include <math.h>

// Log-NCDE persistent single-WGP kernel for MI455X (gfx1250).
// Strictly sequential RK4 scan -> latency bound; all weights resident in LDS
// as f16; every matmul phase uses v_wmma_f32_16x16x32_f16 (wave32 WMMA).
// This revision: one shared exponential per softplus+sigmoid pair, raw
// v_rcp_f32 instead of precise IEEE divide on the critical path.

#define DD    6
#define ST    64
#define HD    128
#define OUTD  10
#define TT    257
#define WW    8
#define NWW   32
#define NSTEP 256

typedef __attribute__((ext_vector_type(16))) _Float16 v16h;
typedef __attribute__((ext_vector_type(8)))  _Float16 v8h;
typedef __attribute__((ext_vector_type(8)))  float    v8f;

struct alignas(16) Smem {
  _Float16 W1h[HD * ST];      // vf layer1 weights, row-major (ldk=64)
  _Float16 W2h[HD * HD];      // vf layer2 weights (ldk=128)
  _Float16 W3h[DD * ST * HD]; // vf layer3 weights 384x128 (ldk=128)
  _Float16 Xbuf[2][16 * HD];  // ping-pong B operands, col-major [col][row]
  float b1f[HD], b2f[HD], b3f[DD * ST];
  float sig1[HD], sig2[HD];        // softplus' = sigmoid of pre-activations
  float Ff[DD * ST], tpf[DD * ST]; // F = tanh(pre3), tanh' = 1-F^2
  float lin[ST];                   // Fm^T @ av
  float diag[DD * ST];             // tanh'-scaled block-diag of W3@T2@M^T
  float kacc[ST], ycur[ST];
  float effa[NWW][DD];             // a / dtw per window
  float effA[NWW][DD * DD];        // 0.5*(S-S^T)/dtw per window
  float dts[NSTEP];
  float Wr[OUTD * ST], rb[OUTD];
  float tmpa[HD], tmpb[HD];        // init-MLP temps
};

// ---- fast activation helpers (latency-critical, evaluated 1024x3 times) ----
__device__ inline float fast_exp(float v)  { return __expf(v); }
__device__ inline float fast_rcp(float v) {
#if __has_builtin(__builtin_amdgcn_rcpf)
  return __builtin_amdgcn_rcpf(v);   // raw v_rcp_f32 (arg in [1,2] here)
#else
  return 1.f / v;
#endif
}
__device__ inline float fast_tanh(float v) {
#if defined(__gfx1250__) && __has_builtin(__builtin_amdgcn_tanhf)
  return __builtin_amdgcn_tanhf(v);  // CDNA5 v_tanh_f32
#else
  return tanhf(v);
#endif
}
// softplus(x) and sigmoid(x) from a single exponential:
// t = e^{-|x|}, den = 1+t:  sp = max(x,0)+log(den), sig = x>=0 ? 1/den : t/den
__device__ inline void softplus_sig(float x, float& sp, float& sig) {
  float t = fast_exp(-fabsf(x));
  float den = 1.f + t;
  float r = fast_rcp(den);
  sp = fmaxf(x, 0.f) + __logf(den);
  sig = (x >= 0.f) ? r : t * r;
}
__device__ inline float softplus_f(float v) {
  return fmaxf(v, 0.f) + __logf(1.f + fast_exp(-fabsf(v)));
}

// One 16x16 output tile of  D = W[tile_m*16 .. +15, :] @ X  (f16 in, f32 acc).
// W row-major in LDS with leading dim ldk; X col-major [col][128] in LDS.
__device__ inline v8f wmma_tile(const _Float16* Wl, int ldk, int ksteps,
                                const _Float16* Xb, int tile_m, int lane) {
  v8f acc = {0.f, 0.f, 0.f, 0.f, 0.f, 0.f, 0.f, 0.f};
  const int mrow = tile_m * 16 + (lane & 15);
  const int ncol = lane & 15;
  const int hi   = (lane >> 4) & 1;
#pragma unroll
  for (int ks = 0; ks < ksteps; ++ks) {
    // A fragment (16-bit A 16x32): lanes 0-15 hold K{0..7,16..23},
    // lanes 16-31 hold K{8..15,24..31} of the same row.
    const _Float16* pa = Wl + mrow * ldk + ks * 32 + hi * 8;
    v8h a0 = *(const v8h*)pa;
    v8h a1 = *(const v8h*)(pa + 16);
    v16h av = __builtin_shufflevector(a0, a1, 0, 1, 2, 3, 4, 5, 6, 7,
                                      8, 9, 10, 11, 12, 13, 14, 15);
    // B fragment (16-bit B 32x16): lane n<16 -> K{0..7,16..23} of col n; n+16 -> K{8..15,24..31}.
    const _Float16* pb = Xb + ncol * HD + ks * 32 + hi * 8;
    v8h b0 = *(const v8h*)pb;
    v8h b1 = *(const v8h*)(pb + 16);
    v16h bv = __builtin_shufflevector(b0, b1, 0, 1, 2, 3, 4, 5, 6, 7,
                                      8, 9, 10, 11, 12, 13, 14, 15);
    acc = __builtin_amdgcn_wmma_f32_16x16x32_f16(false, av, false, bv,
                                                 (short)0, acc, false, false);
  }
  return acc;
}

// One drift evaluation. On entry: Xbuf[0] col0 rows 0..63 = ystage (f16),
// all threads synced. On exit (after final barrier): lin[] and diag[] valid.
__device__ void drift_eval(Smem& S, const float* av, const float* Avm,
                           int tid, int lane, int wave) {
  const int ncol = lane & 15;
  const int hi   = (lane >> 4) & 1;

  // Phase A: pre1 = W1 @ y   (M=128, K=64; only col 0 meaningful)
  {
    v8f c = wmma_tile(S.W1h, ST, 2, S.Xbuf[0], wave, lane);
    if (ncol == 0) {
      const int m0 = wave * 16 + hi * 8;
#pragma unroll
      for (int r = 0; r < 8; ++r) {
        float sp, sg;
        softplus_sig(c[r] + S.b1f[m0 + r], sp, sg);
        S.Xbuf[1][m0 + r] = (_Float16)sp; // z1 -> col 0
        S.sig1[m0 + r] = sg;
      }
    }
  }
  __syncthreads();

  // Phase B: pre2 = W2 @ z1  (M=128, K=128)
  {
    v8f c = wmma_tile(S.W2h, HD, 4, S.Xbuf[1], wave, lane);
    if (ncol == 0) {
      const int m0 = wave * 16 + hi * 8;
#pragma unroll
      for (int r = 0; r < 8; ++r) {
        float sp, sg;
        softplus_sig(c[r] + S.b2f[m0 + r], sp, sg);
        S.Xbuf[0][m0 + r] = (_Float16)sp; // z2 -> col 0
        S.sig2[m0 + r] = sg;
      }
    }
  }
  __syncthreads();

  // Phase C: F = tanh(W3 @ z2 + b3)  (M=384, K=128)
#pragma unroll
  for (int t = 0; t < 3; ++t) {
    const int tile = wave * 3 + t;
    v8f c = wmma_tile(S.W3h, HD, 4, S.Xbuf[0], tile, lane);
    if (ncol == 0) {
      const int m0 = tile * 16 + hi * 8;
#pragma unroll
      for (int r = 0; r < 8; ++r) {
        float f = fast_tanh(c[r] + S.b3f[m0 + r]);
        S.Ff[m0 + r] = f;
        S.tpf[m0 + r] = 1.f - f * f;
      }
    }
  }
  __syncthreads();

  // lin[k] = sum_i F[i,k]*av[i];  M[j,l] = sum_i Av[i,j]*F[i,l] -> Xbuf[1] cols 0..5
#pragma unroll
  for (int w = tid; w < ST + DD * ST; w += 256) {
    if (w < ST) {
      float acc = 0.f;
#pragma unroll
      for (int i = 0; i < DD; ++i) acc += S.Ff[i * ST + w] * av[i];
      S.lin[w] = acc;
    } else {
      const int idx = w - ST;
      const int j = idx >> 6, l = idx & 63;
      float acc = 0.f;
#pragma unroll
      for (int i = 0; i < DD; ++i) acc += Avm[i * DD + j] * S.Ff[i * ST + l];
      S.Xbuf[1][j * HD + l] = (_Float16)acc;
    }
  }
  __syncthreads();

  // Phase D: G = diag(sig1) * (W1 @ M^T)   (M=128, K=64, 6 live cols)
  {
    v8f c = wmma_tile(S.W1h, ST, 2, S.Xbuf[1], wave, lane);
    const int m0 = wave * 16 + hi * 8;
    v8h o;
#pragma unroll
    for (int r = 0; r < 8; ++r) o[r] = (_Float16)(c[r] * S.sig1[m0 + r]);
    *(v8h*)(&S.Xbuf[0][ncol * HD + m0]) = o;
  }
  __syncthreads();

  // Phase E: H = diag(sig2) * (W2 @ G)     (M=128, K=128)
  {
    v8f c = wmma_tile(S.W2h, HD, 4, S.Xbuf[0], wave, lane);
    const int m0 = wave * 16 + hi * 8;
    v8h o;
#pragma unroll
    for (int r = 0; r < 8; ++r) o[r] = (_Float16)(c[r] * S.sig2[m0 + r]);
    *(v8h*)(&S.Xbuf[1][ncol * HD + m0]) = o;
  }
  __syncthreads();

  // Phase F: U = W3 @ H; keep tanh'-scaled block-diagonal (exactly one writer)
#pragma unroll
  for (int t = 0; t < 3; ++t) {
    const int tile = wave * 3 + t;
    v8f c = wmma_tile(S.W3h, HD, 4, S.Xbuf[1], tile, lane);
    const int m0 = tile * 16 + hi * 8;
#pragma unroll
    for (int r = 0; r < 8; ++r) {
      const int m = m0 + r;
      if (ncol == (m >> 6)) S.diag[m] = S.tpf[m] * c[r];
    }
  }
  __syncthreads();
}

// drift result for state index k (call with tid < ST, after drift_eval)
__device__ inline float drift_val(Smem& S, int k) {
  float d = S.lin[k];
#pragma unroll
  for (int j = 0; j < DD; ++j) d += S.diag[j * ST + k];
  return d;
}

__global__ void __launch_bounds__(256, 1)
logncde_kernel(const float* __restrict__ ts, const float* __restrict__ x,
               const float* __restrict__ iW0, const float* __restrict__ ib0,
               const float* __restrict__ iW1, const float* __restrict__ ib1,
               const float* __restrict__ iW2, const float* __restrict__ ib2,
               const float* __restrict__ vW0, const float* __restrict__ vb0,
               const float* __restrict__ vW1, const float* __restrict__ vb1,
               const float* __restrict__ vW2, const float* __restrict__ vb2,
               const float* __restrict__ Wr, const float* __restrict__ rb,
               float* __restrict__ out) {
  extern __shared__ char smem_raw[];
  Smem& S = *reinterpret_cast<Smem*>(smem_raw);
  const int tid = threadIdx.x;
  const int lane = tid & 31;
  const int wave = tid >> 5;

  // ---- one-time: stage weights (f16) + biases + readout into LDS ----
  for (int i = tid; i < HD * ST; i += 256) S.W1h[i] = (_Float16)vW0[i];
  for (int i = tid; i < HD * HD; i += 256) S.W2h[i] = (_Float16)vW1[i];
  for (int i = tid; i < DD * ST * HD; i += 256) S.W3h[i] = (_Float16)vW2[i];
  for (int i = tid; i < HD; i += 256) { S.b1f[i] = vb0[i]; S.b2f[i] = vb1[i]; }
  for (int i = tid; i < DD * ST; i += 256) S.b3f[i] = vb2[i];
  for (int i = tid; i < OUTD * ST; i += 256) S.Wr[i] = Wr[i];
  for (int i = tid; i < OUTD; i += 256) S.rb[i] = rb[i];
  for (int s = tid; s < NSTEP; s += 256) S.dts[s] = ts[s + 1] - ts[s];

  // ---- one-time: per-window log-signature terms (a, A)/dtw ----
  if (tid < NWW) {
    const int n = tid;
    float dXl[WW][DD];
    for (int k = 0; k < WW; ++k)
      for (int i = 0; i < DD; ++i)
        dXl[k][i] = x[(n * WW + k + 1) * DD + i] - x[(n * WW + k) * DD + i];
    const float dtw = ts[(n + 1) * WW] - ts[n * WW];
    float P[DD], a[DD], Sm[DD][DD];
    for (int i = 0; i < DD; ++i) { P[i] = 0.f; a[i] = 0.f; }
    for (int i = 0; i < DD; ++i) for (int j = 0; j < DD; ++j) Sm[i][j] = 0.f;
    for (int k = 0; k < WW; ++k) {
      for (int i = 0; i < DD; ++i)
        for (int j = 0; j < DD; ++j) Sm[i][j] += P[i] * dXl[k][j]; // exclusive prefix
      for (int i = 0; i < DD; ++i) { a[i] += dXl[k][i]; P[i] += dXl[k][i]; }
    }
    for (int i = 0; i < DD; ++i) S.effa[n][i] = a[i] / dtw;
    for (int i = 0; i < DD; ++i)
      for (int j = 0; j < DD; ++j)
        S.effA[n][i * DD + j] = 0.5f * (Sm[i][j] - Sm[j][i]) / dtw;
  }
  __syncthreads();

  // ---- one-time: h0 = init_mlp(x[0]) in fp32 (tiny GEMVs) ----
  if (tid < HD) {
    float acc = ib0[tid];
#pragma unroll
    for (int i = 0; i < DD; ++i) acc += iW0[tid * DD + i] * x[i];
    S.tmpa[tid] = softplus_f(acc);
  }
  __syncthreads();
  if (tid < HD) {
    float acc = ib1[tid];
    for (int j = 0; j < HD; ++j) acc += iW1[tid * HD + j] * S.tmpa[j];
    S.tmpb[tid] = softplus_f(acc);
  }
  __syncthreads();
  if (tid < ST) {
    float acc = ib2[tid];
    for (int j = 0; j < HD; ++j) acc += iW2[tid * HD + j] * S.tmpb[j];
    S.ycur[tid] = acc;
  }
  __syncthreads();
  if (tid < OUTD) { // row 0 readout
    float a0 = 0.f, a1 = 0.f, a2 = 0.f, a3 = 0.f;
#pragma unroll
    for (int k = 0; k < ST; k += 4) {
      a0 += S.Wr[tid * ST + k + 0] * S.ycur[k + 0];
      a1 += S.Wr[tid * ST + k + 1] * S.ycur[k + 1];
      a2 += S.Wr[tid * ST + k + 2] * S.ycur[k + 2];
      a3 += S.Wr[tid * ST + k + 3] * S.ycur[k + 3];
    }
    out[tid] = S.rb[tid] + ((a0 + a1) + (a2 + a3));
  }
  if (tid < ST) S.Xbuf[0][tid] = (_Float16)S.ycur[tid]; // stage-1 operand
  __syncthreads();

  // ---- sequential RK4 scan (the latency-bound core) ----
  for (int s = 0; s < NSTEP; ++s) {
    const float dt = S.dts[s];
    const float* av  = S.effa[s >> 3];
    const float* Avm = S.effA[s >> 3];

    // stage 1: k1 = f(y); next operand y + dt/2*k1
    drift_eval(S, av, Avm, tid, lane, wave);
    if (tid < ST) {
      float d = drift_val(S, tid);
      S.kacc[tid] = d;
      S.Xbuf[0][tid] = (_Float16)(S.ycur[tid] + 0.5f * dt * d);
    }
    __syncthreads();

    // stage 2: k2; next operand y + dt/2*k2
    drift_eval(S, av, Avm, tid, lane, wave);
    if (tid < ST) {
      float d = drift_val(S, tid);
      S.kacc[tid] += 2.f * d;
      S.Xbuf[0][tid] = (_Float16)(S.ycur[tid] + 0.5f * dt * d);
    }
    __syncthreads();

    // stage 3: k3; next operand y + dt*k3
    drift_eval(S, av, Avm, tid, lane, wave);
    if (tid < ST) {
      float d = drift_val(S, tid);
      S.kacc[tid] += 2.f * d;
      S.Xbuf[0][tid] = (_Float16)(S.ycur[tid] + dt * d);
    }
    __syncthreads();

    // stage 4: k4; y <- y + dt/6*(k1+2k2+2k3+k4)
    drift_eval(S, av, Avm, tid, lane, wave);
    if (tid < ST) {
      float d = drift_val(S, tid);
      S.ycur[tid] += (dt / 6.f) * (S.kacc[tid] + d);
    }
    __syncthreads();

    // readout row s+1 and stage-1 operand of next step
    if (tid < OUTD) {
      float a0 = 0.f, a1 = 0.f, a2 = 0.f, a3 = 0.f;
#pragma unroll
      for (int k = 0; k < ST; k += 4) {
        a0 += S.Wr[tid * ST + k + 0] * S.ycur[k + 0];
        a1 += S.Wr[tid * ST + k + 1] * S.ycur[k + 1];
        a2 += S.Wr[tid * ST + k + 2] * S.ycur[k + 2];
        a3 += S.Wr[tid * ST + k + 3] * S.ycur[k + 3];
      }
      out[(s + 1) * OUTD + tid] = S.rb[tid] + ((a0 + a1) + (a2 + a3));
    }
    if (tid < ST) S.Xbuf[0][tid] = (_Float16)S.ycur[tid];
    __syncthreads();
  }
}

extern "C" void kernel_launch(void* const* d_in, const int* in_sizes, int n_in,
                              void* d_out, int out_size, void* d_ws, size_t ws_size,
                              hipStream_t stream) {
  (void)in_sizes; (void)n_in; (void)out_size; (void)d_ws; (void)ws_size;
  const float* ts  = (const float*)d_in[0];
  const float* x   = (const float*)d_in[1];
  const float* iW0 = (const float*)d_in[2];
  const float* ib0 = (const float*)d_in[3];
  const float* iW1 = (const float*)d_in[4];
  const float* ib1 = (const float*)d_in[5];
  const float* iW2 = (const float*)d_in[6];
  const float* ib2 = (const float*)d_in[7];
  const float* vW0 = (const float*)d_in[8];
  const float* vb0 = (const float*)d_in[9];
  const float* vW1 = (const float*)d_in[10];
  const float* vb1 = (const float*)d_in[11];
  const float* vW2 = (const float*)d_in[12];
  const float* vb2 = (const float*)d_in[13];
  const float* Wr  = (const float*)d_in[14];
  const float* rb  = (const float*)d_in[15];
  float* out = (float*)d_out;

  static_assert(sizeof(Smem) <= 300 * 1024, "fits in 320KB WGP LDS");
  (void)hipFuncSetAttribute((const void*)logncde_kernel,
                            hipFuncAttributeMaxDynamicSharedMemorySize,
                            (int)sizeof(Smem));
  // Single persistent workgroup: the scan is sequential; 8 wave32 waves
  // cooperate on each drift's six WMMA phases with all state in LDS.
  logncde_kernel<<<1, 256, sizeof(Smem), stream>>>(
      ts, x, iW0, ib0, iW1, ib1, iW2, ib2,
      vW0, vb0, vW1, vb1, vW2, vb2, Wr, rb, out);
}
